// AttentionModule_38113539784744
// MI455X (gfx1250) — matrix operand order repaired and checked by reading, not compiled, not run on hardware
//
#include <hip/hip_runtime.h>

typedef __attribute__((ext_vector_type(16))) _Float16 v16h;
typedef __attribute__((ext_vector_type(8)))  _Float16 v8h;
typedef __attribute__((ext_vector_type(8)))  float    v8f;
typedef __attribute__((ext_vector_type(4)))  float    f32x4;

#define S_LEN 2048
#define D_DIM 64
#define NWAVES 8      // 256 threads = 8 wave32
#define QT_PER_BLK 8  // 8 query tiles of 16 rows per block (same head)

__global__ __launch_bounds__(256)
void fa_fwd_causal_coop(const float* __restrict__ Q,
                        const float* __restrict__ K,
                        const float* __restrict__ V,
                        float* __restrict__ O)
{
    // double-buffered shared KV tiles (f16), V stored transposed [d][kv]
    __shared__ _Float16 Ksh[2][32 * 64];
    __shared__ _Float16 Vsh[2][64 * 32];
    // per-wave P scratch for C/D-layout -> A-layout conversion
    __shared__ _Float16 Psh[NWAVES][16 * 32];

    const int tid  = threadIdx.x;
    const int lane = tid & 31;
    const int wave = tid >> 5;

    const int bh     = blockIdx.x >> 4;          // b*H + h (64 heads)
    const int qgrp   = blockIdx.x & 15;          // 16 groups of 8 q-tiles
    const int qbase  = (qgrp * QT_PER_BLK + wave) * 16;

    const int hs = lane >> 4;                    // lane half select
    const int ln = lane & 15;

    const float* Qp = Q + (size_t)bh * S_LEN * D_DIM;
    const float* Kp = K + (size_t)bh * S_LEN * D_DIM;
    const float* Vp = V + (size_t)bh * S_LEN * D_DIM;
    float*       Op = O + (size_t)bh * S_LEN * D_DIM;

    // block-wide kv tile count = cutoff of the last (largest) q tile in block
    const int n_kt = ((qgrp * QT_PER_BLK + 7) * 16 + 15) / 32 + 1;

    // ---- cooperative KV tile fill: 256 threads x 8 f32 per matrix ----
    const int frow = tid >> 3;          // kv row 0..31
    const int fcol = (tid & 7) * 8;     // d start 0,8,..,56
    auto fill = [&](int kt, int buf) {
        const float* kr = Kp + (size_t)(kt * 32 + frow) * D_DIM + fcol;
        const float* vr = Vp + (size_t)(kt * 32 + frow) * D_DIM + fcol;
        f32x4 k0 = *(const f32x4*)kr;
        f32x4 k1 = *(const f32x4*)(kr + 4);
        f32x4 v0 = *(const f32x4*)vr;
        f32x4 v1 = *(const f32x4*)(vr + 4);
        v8h kh, vh;
#pragma unroll
        for (int e = 0; e < 4; ++e) {
            kh[e]     = (_Float16)k0[e];
            kh[e + 4] = (_Float16)k1[e];
            vh[e]     = (_Float16)v0[e];
            vh[e + 4] = (_Float16)v1[e];
        }
        *(v8h*)(&Ksh[buf][frow * 64 + fcol]) = kh;      // one 16B ds store
#pragma unroll
        for (int e = 0; e < 8; ++e)                      // transposed scatter
            Vsh[buf][(fcol + e) * 32 + frow] = vh[e];
    };

    // ---- Q tile as two A-fragments (contraction d = 0..31 and 32..63) ----
    v16h qa[2];
    {
        const float* qrow = Qp + (size_t)(qbase + ln) * D_DIM;
        const int kb = hs * 8;
#pragma unroll
        for (int c = 0; c < 2; ++c)
#pragma unroll
            for (int e = 0; e < 8; ++e) {
                qa[c][e]     = (_Float16)qrow[c * 32 + kb + e];
                qa[c][e + 8] = (_Float16)qrow[c * 32 + 16 + kb + e];
            }
    }

    // constant B-fragment with an all-ones column 0: row-sum of P via WMMA
    v16h onesf;
    {
        const _Float16 o1 = (ln == 0) ? (_Float16)1.0f : (_Float16)0.0f;
#pragma unroll
        for (int e = 0; e < 16; ++e) onesf[e] = o1;
    }

    v8f oacc[4] = {};
    v8f lacc    = {};            // unnormalized row sums (valid in lanes ln==0)
    float m_i[8];
#pragma unroll
    for (int v = 0; v < 8; ++v) m_i[v] = -__builtin_inff();

    // scores kept in log2 domain: scale = log2(e)/sqrt(64)
    const float scale2 = 0.125f * 1.44269504088896340736f;

    fill(0, 0);

    for (int kt = 0; kt < n_kt; ++kt) {
        __syncthreads();   // buf[kt&1] ready; everyone done with buf[(kt+1)&1]

        if (kt + 1 < n_kt) fill(kt + 1, (kt + 1) & 1);
        if (kt + 2 < n_kt) {
            __builtin_prefetch(Kp + (size_t)((kt + 2) * 32 + frow) * D_DIM + fcol, 0, 1);
            __builtin_prefetch(Vp + (size_t)((kt + 2) * 32 + frow) * D_DIM + fcol, 0, 1);
        }

        const int kvbase = kt * 32;
        if (kvbase >= qbase + 16) continue;   // fully above this wave's diagonal

        const _Float16* Kb = Ksh[kt & 1];
        const _Float16* Vt = Vsh[kt & 1];

        // ---- S = Q K^T : two 16x16 score tiles ----
        v8f sacc[2];
#pragma unroll
        for (int j = 0; j < 2; ++j) {
            v8f c = {};
#pragma unroll
            for (int ch = 0; ch < 2; ++ch) {
                // B-fragment: col n = kv (16j+ln); k = d = ch*32 + 8*hs + e (e<8), ch*32 + 16 + 8*hs + (e-8) (e>=8)
                const _Float16* kp = Kb + (16 * j + ln) * 64 + ch * 32 + hs * 8;
                v8h klo = *(const v8h*)(kp);
                v8h khi = *(const v8h*)(kp + 16);
                v16h kf;
#pragma unroll
                for (int e = 0; e < 8; ++e) { kf[e] = klo[e]; kf[e + 8] = khi[e]; }
                c = __builtin_amdgcn_wmma_f32_16x16x32_f16(
                        false, qa[ch], false, kf, (short)0, c, false, false);
            }
            sacc[j] = c;
        }

        // ---- scale (log2 domain) + causal mask (last tile only) ----
        const bool need_mask = (kvbase + 31 > qbase);
#pragma unroll
        for (int j = 0; j < 2; ++j) {
            const int kvg = kvbase + 16 * j + ln;
#pragma unroll
            for (int v = 0; v < 8; ++v) {
                float s = sacc[j][v] * scale2;
                if (need_mask && (kvg > qbase + v + 8 * hs)) s = -__builtin_inff();
                sacc[j][v] = s;
            }
        }

        // ---- online softmax: row max via 16-lane shuffles, exp2 probabilities ----
        float pex[2][8];
#pragma unroll
        for (int v = 0; v < 8; ++v) {
            float rm = fmaxf(sacc[0][v], sacc[1][v]);
#pragma unroll
            for (int sh = 1; sh < 16; sh <<= 1)
                rm = fmaxf(rm, __shfl_xor(rm, sh, 32));
            const float m_new = fmaxf(m_i[v], rm);
            const float corr  = __builtin_amdgcn_exp2f(m_i[v] - m_new); // v_exp_f32
            m_i[v] = m_new;
            pex[0][v] = __builtin_amdgcn_exp2f(sacc[0][v] - m_new);
            pex[1][v] = __builtin_amdgcn_exp2f(sacc[1][v] - m_new);
            // rescale running accumulators (O chunks + row-sum column)
#pragma unroll
            for (int cc = 0; cc < 4; ++cc)
                oacc[cc][v] *= corr;
            lacc[v] *= corr;
        }

        // ---- P: C/D layout -> A layout via per-wave LDS roundtrip ----
        _Float16* pl = Psh[wave];
#pragma unroll
        for (int j = 0; j < 2; ++j)
#pragma unroll
            for (int v = 0; v < 8; ++v)
                pl[(v + 8 * hs) * 32 + 16 * j + ln] = (_Float16)pex[j][v];

        v16h pa;
        {
            const int kb = hs * 16;
            v8h lo = *(const v8h*)(pl + ln * 32 + kb);        // ds_load_b128
            v8h hi = *(const v8h*)(pl + ln * 32 + 8 + kb);    // ds_load_b128
#pragma unroll
            for (int e = 0; e < 8; ++e) { pa[e] = lo[e]; pa[e + 8] = hi[e]; }
        }

        // ---- O += P V (4 WMMAs) and row-sum += P * ones (1 WMMA) ----
#pragma unroll
        for (int cc = 0; cc < 4; ++cc) {
            // B-fragment from transposed V: n = d (cc*16+ln); k = 16*hs + e
            v16h vf = *(const v16h*)(Vt + (cc * 16 + ln) * 32 + 16 * hs);
            oacc[cc] = __builtin_amdgcn_wmma_f32_16x16x32_f16(
                           false, pa, false, vf, (short)0, oacc[cc], false, false);
        }
        lacc = __builtin_amdgcn_wmma_f32_16x16x32_f16(
                   false, pa, false, onesf, (short)0, lacc, false, false);
    }

    // ---- normalize by row sums (broadcast from lane 16*hs) and store ----
#pragma unroll
    for (int v = 0; v < 8; ++v) {
        const float lsum = __shfl(lacc[v], 16 * hs, 32);
        const float inv  = (lsum > 0.0f) ? 1.0f / lsum : 0.0f;
        float* orow = Op + (size_t)(qbase + v + 8 * hs) * D_DIM;
#pragma unroll
        for (int cc = 0; cc < 4; ++cc)
            orow[cc * 16 + ln] = oacc[cc][v] * inv;
    }
}

extern "C" void kernel_launch(void* const* d_in, const int* in_sizes, int n_in,
                              void* d_out, int out_size, void* d_ws, size_t ws_size,
                              hipStream_t stream) {
    const float* Q = (const float*)d_in[0];
    const float* K = (const float*)d_in[1];
    const float* V = (const float*)d_in[2];
    // d_in[3] (causal bool mask) is implemented analytically.
    float* O = (float*)d_out;

    // 64 heads x 16 q-groups = 1024 blocks of 256 threads (8 wave32 each)
    const int blocks = 64 * (S_LEN / (16 * QT_PER_BLK));
    fa_fwd_causal_coop<<<blocks, 256, 0, stream>>>(Q, K, V, O);
}
